// ECDiTRoutingLayer_19713899888758
// MI455X (gfx1250) — matrix operand order repaired and checked by reading, not compiled, not run on hardware
//
#include <hip/hip_runtime.h>
#include <hip/hip_bf16.h>

// ---------------------------------------------------------------------------
// Problem constants (from reference)
// ---------------------------------------------------------------------------
#define DIMD   1024
#define NEXP   8
#define HID    4096
#define BB     4
#define SS     2048
#define CAP    513      // int(2048*2.0/8)+1
#define MPAD   576      // CAP padded to 9 tiles of 64
#define BE     32       // B*E

#define LDSK   40       // 32 K shorts + 8 pad -> 80B row stride (16B aligned)

typedef __attribute__((ext_vector_type(16))) __bf16 v16bf;
typedef __attribute__((ext_vector_type(8)))  float  v8f;
typedef __attribute__((ext_vector_type(4)))  int    v4i;

// async global->LDS builtin takes v4i* in AS(1)/AS(3)
typedef __attribute__((address_space(1))) v4i* gptr128;
typedef __attribute__((address_space(3))) v4i* lptr128;

union AFrag { v16bf v; unsigned int u[8]; };

#if defined(__HIP_DEVICE_COMPILE__) && __has_builtin(__builtin_amdgcn_global_load_async_to_lds_b128) && __has_builtin(__builtin_amdgcn_s_wait_asynccnt)
#define USE_ASYNC_LDS 1
#else
#define USE_ASYNC_LDS 0
#endif

__device__ __forceinline__ unsigned short f2bf(float f) {
    union { float f; unsigned u; } x; x.f = f;
    unsigned r = x.u + 0x7FFFu + ((x.u >> 16) & 1u);   // round-to-nearest-even
    return (unsigned short)(r >> 16);
}

// pack two fp32 -> packed bf16 dword (lo in [15:0], hi in [31:16])
__device__ __forceinline__ unsigned int pk_bf16(float lo, float hi) {
#if defined(__HIP_DEVICE_COMPILE__) && __has_builtin(__builtin_amdgcn_cvt_pk_bf16_f32)
    auto r = __builtin_amdgcn_cvt_pk_bf16_f32(lo, hi);   // V_CVT_PK_BF16_F32
    unsigned int u; __builtin_memcpy(&u, &r, 4); return u;
#else
    return (unsigned)f2bf(lo) | ((unsigned)f2bf(hi) << 16);
#endif
}

__device__ __forceinline__ void wait_async_all() {
#if USE_ASYNC_LDS
    __builtin_amdgcn_s_wait_asynccnt(0);
#endif
}

// ---------------------------------------------------------------------------
// Kernel 1: router logits + softmax.  One wave per token.
// ---------------------------------------------------------------------------
__global__ __launch_bounds__(256) void router_kernel(const float* __restrict__ x,
                                                     const float* __restrict__ rw,
                                                     float* __restrict__ scores) {
    const int wave = threadIdx.x >> 5, lane = threadIdx.x & 31;
    const int t = blockIdx.x * 8 + wave;              // t in [0, B*S)
    const float* xr = x + (size_t)t * DIMD;
    float acc[NEXP];
#pragma unroll
    for (int e = 0; e < NEXP; ++e) acc[e] = 0.f;
    for (int i = 0; i < DIMD / 32; ++i) {
        const int d = lane + i * 32;
        const float xv = xr[d];
#pragma unroll
        for (int e = 0; e < NEXP; ++e) acc[e] += xv * rw[e * DIMD + d];
    }
#pragma unroll
    for (int e = 0; e < NEXP; ++e)
        for (int off = 16; off; off >>= 1) acc[e] += __shfl_xor(acc[e], off, 32);
    if (lane == 0) {
        float mx = acc[0];
#pragma unroll
        for (int e = 1; e < NEXP; ++e) mx = fmaxf(mx, acc[e]);
        float s = 0.f, ex[NEXP];
#pragma unroll
        for (int e = 0; e < NEXP; ++e) { ex[e] = expf(acc[e] - mx); s += ex[e]; }
        const float inv = 1.f / s;
#pragma unroll
        for (int e = 0; e < NEXP; ++e) scores[(size_t)t * NEXP + e] = ex[e] * inv;
    }
}

// ---------------------------------------------------------------------------
// Kernel 2: per-(b,e) top-513 of 2048 via LDS bitonic sort of packed keys.
// key = score_bits<<32 | (0xFFFFFFFF - idx)  (scores > 0 => bits are ordered)
// ---------------------------------------------------------------------------
__global__ __launch_bounds__(1024) void topk_kernel(const float* __restrict__ scores,
                                                    float* __restrict__ gates,
                                                    int* __restrict__ sel) {
    __shared__ unsigned long long K[SS];
    const int be = blockIdx.x, b = be >> 3, e = be & 7;
    const int tid = threadIdx.x;
    for (int i = tid; i < SS; i += 1024) {
        const float s = scores[((size_t)(b * SS + i)) * NEXP + e];
        K[i] = ((unsigned long long)__float_as_uint(s) << 32) |
               (unsigned)(0xFFFFFFFFu - (unsigned)i);
    }
    for (int kk = 2; kk <= SS; kk <<= 1)
        for (int j = kk >> 1; j > 0; j >>= 1) {
            __syncthreads();
            for (int i = tid; i < SS; i += 1024) {
                const int ixj = i ^ j;
                if (ixj > i) {
                    const unsigned long long a = K[i], c = K[ixj];
                    const bool desc = ((i & kk) == 0);
                    if (desc ? (a < c) : (a > c)) { K[i] = c; K[ixj] = a; }
                }
            }
        }
    __syncthreads();
    if (tid < MPAD) {
        if (tid < CAP) {
            const unsigned long long kv = K[tid];
            gates[be * MPAD + tid] = __uint_as_float((unsigned)(kv >> 32));
            sel[be * MPAD + tid]   = (int)(0xFFFFFFFFu - (unsigned)kv);
        } else {
            gates[be * MPAD + tid] = 0.f;
            sel[be * MPAD + tid]   = 0;
        }
    }
}

// ---------------------------------------------------------------------------
// WMMA fragment loaders (bf16 16x16x32).  LDS tiles keep K contiguous so
// every K-pair is one dword ds_load.
//   A frag: lane 0-15 row M=l, K = {0..7,16..23}; lane 16-31 K = {8..15,24..31}
//   B frag: lane 0-15 col N=l, K = {0..7,16..23}; lane 16-31 K = {8..15,24..31}
//   C/D:    lane<16 -> M=r, N=lane; lane>=16 -> M=8+r, N=lane-16
// ---------------------------------------------------------------------------
__device__ __forceinline__ void load_afrag(AFrag& f, const unsigned short (*T)[LDSK],
                                           int row, int lhalf) {
#pragma unroll
    for (int v = 0; v < 8; ++v) {
        const int kk = (v < 4) ? (lhalf * 8 + 2 * v) : (16 + lhalf * 8 + 2 * (v - 4));
        f.u[v] = *(const unsigned int*)&T[row][kk];
    }
}
__device__ __forceinline__ void load_bfrag(AFrag& f, const unsigned short (*T)[LDSK],
                                           int col, int lhalf) {
#pragma unroll
    for (int v = 0; v < 8; ++v) {
        const int kk = (v < 4) ? (lhalf * 8 + 2 * v) : (16 + lhalf * 8 + 2 * (v - 4));
        f.u[v] = *(const unsigned int*)&T[col][kk];
    }
}

// ---------------------------------------------------------------------------
// Kernel 3: GEMM1 + bias + exact GELU.  A = gathered tokens (fp32->bf16),
// B = w1[e] (fp32->bf16, transposed into LDS).  Out: H bf16 [be][MPAD][HID].
// Workgroup: 256 thr = 8 waves (2M x 4N), tile 64x256, K-step 32,
// double-buffered LDS.  Wave tile 32x64 -> 8 WMMA per 6 frag loads.
// ---------------------------------------------------------------------------
__global__ __launch_bounds__(256) void gemm1_kernel(const float* __restrict__ x,
                                                    const float* __restrict__ w1,
                                                    const float* __restrict__ b1,
                                                    const int* __restrict__ sel,
                                                    unsigned short* __restrict__ Hbuf) {
    const int be = blockIdx.z, b = be >> 3, e = be & 7;
    const int m0 = blockIdx.y * 64;
    const int n0 = blockIdx.x * 256;
    __shared__ __align__(16) unsigned short As[2][64][LDSK];
    __shared__ __align__(16) unsigned short Bs[2][256][LDSK];

    const int tid = threadIdx.x;
    const int lane = tid & 31, wave = tid >> 5;
    const int wm = wave >> 2, wn = wave & 3;
    const int l15 = lane & 15, lhalf = lane >> 4;

    // A gather pointers: 512 float4 items, 2 per thread, fixed (m,kq) mapping
    const float* arow[2]; int am[2], akq[2];
#pragma unroll
    for (int t2 = 0; t2 < 2; ++t2) {
        const int it = tid + t2 * 256;
        am[t2] = it >> 3; akq[t2] = it & 7;
        const int tok = sel[be * MPAD + m0 + am[t2]];
        arow[t2] = x + ((size_t)(b * SS + tok) << 10) + akq[t2] * 4;
    }

    auto stageA = [&](int sb, int k0) {
#pragma unroll
        for (int t2 = 0; t2 < 2; ++t2) {
            const float4 v = *(const float4*)(arow[t2] + k0);
            *(uint2*)&As[sb][am[t2]][akq[t2] * 4] =
                make_uint2(pk_bf16(v.x, v.y), pk_bf16(v.z, v.w));
        }
    };
    auto stageB = [&](int sb, int k0) {
#pragma unroll
        for (int t2 = 0; t2 < 4; ++t2) {            // 1024 pair items, 4/thread
            const int it = tid + t2 * 256;
            const int kp = it >> 6, nq = it & 63;   // K pair, 4-wide N chunk
            const float* s0 = w1 + ((size_t)e << 22) +
                              (size_t)(k0 + 2 * kp) * HID + n0 + nq * 4;
            const float4 v0 = *(const float4*)s0;
            const float4 v1 = *(const float4*)(s0 + HID);
            *(unsigned int*)&Bs[sb][nq * 4 + 0][2 * kp] = pk_bf16(v0.x, v1.x);
            *(unsigned int*)&Bs[sb][nq * 4 + 1][2 * kp] = pk_bf16(v0.y, v1.y);
            *(unsigned int*)&Bs[sb][nq * 4 + 2][2 * kp] = pk_bf16(v0.z, v1.z);
            *(unsigned int*)&Bs[sb][nq * 4 + 3][2 * kp] = pk_bf16(v0.w, v1.w);
        }
    };

    const v8f z = {0.f, 0.f, 0.f, 0.f, 0.f, 0.f, 0.f, 0.f};
    v8f acc[2][4];
#pragma unroll
    for (int mm = 0; mm < 2; ++mm)
#pragma unroll
        for (int nn = 0; nn < 4; ++nn) acc[mm][nn] = z;

    stageA(0, 0); stageB(0, 0);
    int buf = 0;
    for (int k0 = 0; k0 < DIMD; k0 += 32, buf ^= 1) {
        __syncthreads();
        if (k0 + 32 < DIMD) { stageA(buf ^ 1, k0 + 32); stageB(buf ^ 1, k0 + 32); }
        AFrag af[2], bfr[4];
#pragma unroll
        for (int mm = 0; mm < 2; ++mm)
            load_afrag(af[mm], As[buf], wm * 32 + mm * 16 + l15, lhalf);
#pragma unroll
        for (int nn = 0; nn < 4; ++nn)
            load_bfrag(bfr[nn], Bs[buf], wn * 64 + nn * 16 + l15, lhalf);
#pragma unroll
        for (int mm = 0; mm < 2; ++mm)
#pragma unroll
            for (int nn = 0; nn < 4; ++nn)
                acc[mm][nn] = __builtin_amdgcn_wmma_f32_16x16x32_bf16(
                    false, af[mm].v, false, bfr[nn].v, (short)0, acc[mm][nn], false, false);
    }

    // epilogue: bias + exact GELU -> bf16 H
#pragma unroll
    for (int mm = 0; mm < 2; ++mm)
#pragma unroll
        for (int nn = 0; nn < 4; ++nn) {
            const int ng = n0 + wn * 64 + nn * 16 + l15;
            const float bias = b1[e * HID + ng];
#pragma unroll
            for (int r = 0; r < 8; ++r) {
                const int ml = wm * 32 + mm * 16 + lhalf * 8 + r;
                float val = acc[mm][nn][r] + bias;
                val = 0.5f * val * (1.0f + erff(val * 0.70710678118654752f));
                Hbuf[((size_t)be * MPAD + m0 + ml) * HID + ng] = f2bf(val);
            }
        }
}

// ---------------------------------------------------------------------------
// Kernel 4: GEMM2 + bias + gate + atomic scatter-add into out.
// A = H bf16 (async global->LDS copy when available), B = w2[e] (fp32->bf16).
// ---------------------------------------------------------------------------
__global__ __launch_bounds__(256) void gemm2_kernel(const unsigned short* __restrict__ Hbuf,
                                                    const float* __restrict__ w2,
                                                    const float* __restrict__ b2,
                                                    const int* __restrict__ sel,
                                                    const float* __restrict__ gates,
                                                    float* __restrict__ out) {
    const int be = blockIdx.z, b = be >> 3, e = be & 7;
    const int m0 = blockIdx.y * 64;
    const int n0 = blockIdx.x * 256;
    __shared__ __align__(16) unsigned short As[2][64][LDSK];
    __shared__ __align__(16) unsigned short Bs[2][256][LDSK];

    const int tid = threadIdx.x;
    const int lane = tid & 31, wave = tid >> 5;
    const int wm = wave >> 2, wn = wave & 3;
    const int l15 = lane & 15, lhalf = lane >> 4;

    const int amr = tid >> 2, ack = (tid & 3) * 8;   // 64 rows x 4 16B chunks
    const unsigned short* asrc = Hbuf + ((size_t)be * MPAD + m0 + amr) * HID + ack;

    auto stageA = [&](int sb, int k0) {
#if USE_ASYNC_LDS
        __builtin_amdgcn_global_load_async_to_lds_b128(
            (gptr128)(asrc + k0),
            (lptr128)&As[sb][amr][ack],
            0, 0);
#else
        *(uint4*)&As[sb][amr][ack] = *(const uint4*)(asrc + k0);
#endif
    };
    auto stageB = [&](int sb, int k0) {
#pragma unroll
        for (int t2 = 0; t2 < 4; ++t2) {
            const int it = tid + t2 * 256;
            const int kp = it >> 6, nq = it & 63;
            const float* s0 = w2 + ((size_t)e << 22) +
                              (size_t)(k0 + 2 * kp) * DIMD + n0 + nq * 4;
            const float4 v0 = *(const float4*)s0;
            const float4 v1 = *(const float4*)(s0 + DIMD);
            *(unsigned int*)&Bs[sb][nq * 4 + 0][2 * kp] = pk_bf16(v0.x, v1.x);
            *(unsigned int*)&Bs[sb][nq * 4 + 1][2 * kp] = pk_bf16(v0.y, v1.y);
            *(unsigned int*)&Bs[sb][nq * 4 + 2][2 * kp] = pk_bf16(v0.z, v1.z);
            *(unsigned int*)&Bs[sb][nq * 4 + 3][2 * kp] = pk_bf16(v0.w, v1.w);
        }
    };

    const v8f z = {0.f, 0.f, 0.f, 0.f, 0.f, 0.f, 0.f, 0.f};
    v8f acc[2][4];
#pragma unroll
    for (int mm = 0; mm < 2; ++mm)
#pragma unroll
        for (int nn = 0; nn < 4; ++nn) acc[mm][nn] = z;

    stageA(0, 0); stageB(0, 0);
    int buf = 0;
    for (int k0 = 0; k0 < HID; k0 += 32, buf ^= 1) {
        wait_async_all();          // this wave's async LDS writes done pre-barrier
        __syncthreads();
        if (k0 + 32 < HID) { stageA(buf ^ 1, k0 + 32); stageB(buf ^ 1, k0 + 32); }
        AFrag af[2], bfr[4];
#pragma unroll
        for (int mm = 0; mm < 2; ++mm)
            load_afrag(af[mm], As[buf], wm * 32 + mm * 16 + l15, lhalf);
#pragma unroll
        for (int nn = 0; nn < 4; ++nn)
            load_bfrag(bfr[nn], Bs[buf], wn * 64 + nn * 16 + l15, lhalf);
#pragma unroll
        for (int mm = 0; mm < 2; ++mm)
#pragma unroll
            for (int nn = 0; nn < 4; ++nn)
                acc[mm][nn] = __builtin_amdgcn_wmma_f32_16x16x32_bf16(
                    false, af[mm].v, false, bfr[nn].v, (short)0, acc[mm][nn], false, false);
    }

    // epilogue: bias, gate, atomic scatter-add (experts may share tokens)
#pragma unroll
    for (int mm = 0; mm < 2; ++mm)
#pragma unroll
        for (int nn = 0; nn < 4; ++nn) {
            const int ng = n0 + wn * 64 + nn * 16 + l15;
            const float bias = b2[e * DIMD + ng];
#pragma unroll
            for (int r = 0; r < 8; ++r) {
                const int c = m0 + wm * 32 + mm * 16 + lhalf * 8 + r;
                if (c < CAP) {
                    const int tok = sel[be * MPAD + c];
                    const float g = gates[be * MPAD + c];
                    atomicAdd(out + ((size_t)(b * SS + tok) << 10) + ng,
                              (acc[mm][nn][r] + bias) * g);
                }
            }
        }
}

// ---------------------------------------------------------------------------
// Launch
// ---------------------------------------------------------------------------
extern "C" void kernel_launch(void* const* d_in, const int* in_sizes, int n_in,
                              void* d_out, int out_size, void* d_ws, size_t ws_size,
                              hipStream_t stream) {
    const float* x  = (const float*)d_in[0];
    const float* rw = (const float*)d_in[1];
    const float* w1 = (const float*)d_in[2];
    const float* b1 = (const float*)d_in[3];
    const float* w2 = (const float*)d_in[4];
    const float* b2 = (const float*)d_in[5];
    float* out = (float*)d_out;

    char* ws = (char*)d_ws;
    const size_t scores_bytes = (size_t)BB * SS * NEXP * sizeof(float);   // 256 KB
    const size_t gates_bytes  = (size_t)BE * MPAD * sizeof(float);        // 72 KB
    const size_t sel_bytes    = (size_t)BE * MPAD * sizeof(int);          // 72 KB
    float* scores = (float*)ws;
    float* gates  = (float*)(ws + scores_bytes);
    int*   sel    = (int*)(ws + scores_bytes + gates_bytes);
    unsigned short* Hbuf = (unsigned short*)(ws + scores_bytes + gates_bytes + sel_bytes);

    (void)hipMemsetAsync(d_out, 0, (size_t)out_size * sizeof(float), stream);

    router_kernel<<<(BB * SS) / 8, 256, 0, stream>>>(x, rw, scores);
    topk_kernel<<<BE, 1024, 0, stream>>>(scores, gates, sel);
    gemm1_kernel<<<dim3(HID / 256, MPAD / 64, BE), 256, 0, stream>>>(x, w1, b1, sel, Hbuf);
    gemm2_kernel<<<dim3(DIMD / 256, MPAD / 64, BE), 256, 0, stream>>>(Hbuf, w2, b2, sel, gates, out);
}
